// self_atten_11321533792298
// MI455X (gfx1250) — matrix operand order repaired and checked by reading, not compiled, not run on hardware
//
#include <hip/hip_runtime.h>
#include <hip/hip_bf16.h>
#include <math.h>

typedef unsigned short u16;
typedef __attribute__((ext_vector_type(16))) __bf16 v16bf;
typedef __attribute__((ext_vector_type(8)))  float  v8f;
typedef __attribute__((ext_vector_type(4)))  unsigned int v4u;

#define T_TOKENS 16384
#define DIMC 1024
#define ADIMC 256
#define HIDC 4096
#define NBATCH 4
#define NTOK 4096
#define SCALE_A 0.25f    /* 16^-0.5 */
#define EPS_LN 1e-6f
#define EPS_NRM 1e-12f

// GEMM tiling
#define BM 128
#define BN 128
#define BK 64
#define LDB_S (BK + 8)   /* LDS row stride (elements), 144B: 16B-aligned, bank-spread */

// ---------- helpers ----------
__device__ __forceinline__ u16 f2bf(float f) {
  unsigned u = __float_as_uint(f);
  unsigned r = u + 0x7FFFu + ((u >> 16) & 1u);   // round-to-nearest-even
  return (u16)(r >> 16);
}

// A-fragment (16x32 bf16, row m striped in-lane):
// lanes 0-15: V0-3 = K 0..7, V4-7 = K 16..23 ; lanes 16-31: K 8..15 / 24..31
__device__ __forceinline__ v16bf load_a_frag(const u16* __restrict__ A, int lda,
                                             int m0, int k0, int r, int half) {
  const u16* p = A + (size_t)(m0 + r) * lda + k0 + half * 8;
  union { v4u u[2]; v16bf v; } un;
  un.u[0] = *(const v4u*)p;        // K base .. base+7
  un.u[1] = *(const v4u*)(p + 16); // K base+16 .. base+23
  return un.v;
}

// B-fragment (32x16 bf16) from N-major LDS tile:
// lane n (0-15) holds column n, V0-3 = K 0..7, V4-7 = K 16..23; lanes 16-31 hold K 8..15 / 24..31
__device__ __forceinline__ v16bf load_b_lds(const u16* Bs, int n_local, int kk,
                                            int r, int half) {
  const u16* p = Bs + (size_t)(n_local + r) * LDB_S + kk + half * 8;
  union { v4u u[2]; v16bf v; } un;
  un.u[0] = *(const v4u*)p;
  un.u[1] = *(const v4u*)(p + 16);
  return un.v;
}

#define WMMA_BF16(A_, B_, C_) \
  __builtin_amdgcn_wmma_f32_16x16x32_bf16(false, (A_), false, (B_), (short)0, (C_), false, false)

// ---------- GEMM: C[M,N] = A[M,K](bf16) * Bt[N,K](bf16) (+bias)(+add) ----------
// OUT_MODE: 0 = f32 store, 1 = bf16 store, 2 = exact-GELU -> bf16 store
template <int OUT_MODE>
__device__ __forceinline__ void store_tile(const v8f& acc, int m0, int n0, int r, int half,
                                           const float* __restrict__ bias,
                                           const float* __restrict__ add,
                                           void* __restrict__ out, int N) {
#pragma unroll
  for (int v = 0; v < 8; ++v) {
    int row = m0 + half * 8 + v;   // C/D layout: VGPR v -> M = half*8 + v
    int col = n0 + r;
    float val = acc[v];
    if (bias) val += bias[col];
    if (add)  val += add[(size_t)row * N + col];
    if (OUT_MODE == 2) val = 0.5f * val * (1.0f + erff(val * 0.70710678118654752f));
    if (OUT_MODE == 0) ((float*)out)[(size_t)row * N + col] = val;
    else               ((u16*)out)[(size_t)row * N + col] = f2bf(val);
  }
}

template <int OUT_MODE>
__global__ __launch_bounds__(256) void gemm_bf16_wmma(
    const u16* __restrict__ A, const u16* __restrict__ Bt,
    const float* __restrict__ bias, const float* __restrict__ add,
    void* __restrict__ out, int M, int N, int K) {
  __shared__ u16 Bs[BN * LDB_S];

  const int tid  = threadIdx.x;
  const int lane = tid & 31;
  const int wave = tid >> 5;
  const int r    = lane & 15;
  const int half = lane >> 4;
  const int wm   = wave & 3;                       // 4 waves along M (32 rows each)
  const int wn   = wave >> 2;                      // 2 waves along N (64 cols each)
  const int m0   = blockIdx.y * BM + wm * 32;
  const int bn0  = blockIdx.x * BN;
  const int n0w  = wn * 64;
  (void)M;

  // cooperative B-stage mapping: thread t loads 32 contiguous elems of one row
  const int snl = tid >> 1;            // 0..127 : local column (row of Bt)
  const int skh = (tid & 1) * 32;      // 0 / 32 : K half

  v8f acc00, acc01, acc02, acc03, acc10, acc11, acc12, acc13;
  const v8f vz = {0.f, 0.f, 0.f, 0.f, 0.f, 0.f, 0.f, 0.f};
  acc00 = acc01 = acc02 = acc03 = acc10 = acc11 = acc12 = acc13 = vz;

  for (int k0 = 0; k0 < K; k0 += BK) {
    // ---- stage B tile [BN x BK] into LDS (N-major, K-contiguous) ----
    {
      const u16* src = Bt + (size_t)(bn0 + snl) * K + k0 + skh;
      u16* dst = &Bs[(size_t)snl * LDB_S + skh];
      const v4u* s4 = (const v4u*)src;
      v4u* d4 = (v4u*)dst;
      d4[0] = s4[0]; d4[1] = s4[1]; d4[2] = s4[2]; d4[3] = s4[3];  // 32 elems
    }
    __syncthreads();

    if (k0 + BK < K) {  // prefetch next K tiles into WGP-level caches
      __builtin_prefetch((const void*)(A  + (size_t)(m0 + r) * K + k0 + BK), 0, 3);
      __builtin_prefetch((const void*)(Bt + (size_t)(bn0 + snl) * K + k0 + BK + skh), 0, 3);
    }

#pragma unroll
    for (int kk = 0; kk < BK; kk += 32) {
      // Materialize ALL fragments first (distinct registers) so the scheduler
      // can issue the loads as one clause, wait once, then run 8 WMMAs
      // back-to-back on the matrix pipe.
      v16bf a0 = load_a_frag(A, K, m0,      k0 + kk, r, half);
      v16bf a1 = load_a_frag(A, K, m0 + 16, k0 + kk, r, half);
      v16bf b0 = load_b_lds(Bs, n0w +  0, kk, r, half);
      v16bf b1 = load_b_lds(Bs, n0w + 16, kk, r, half);
      v16bf b2 = load_b_lds(Bs, n0w + 32, kk, r, half);
      v16bf b3 = load_b_lds(Bs, n0w + 48, kk, r, half);

      acc00 = WMMA_BF16(a0, b0, acc00);
      acc10 = WMMA_BF16(a1, b0, acc10);
      acc01 = WMMA_BF16(a0, b1, acc01);
      acc11 = WMMA_BF16(a1, b1, acc11);
      acc02 = WMMA_BF16(a0, b2, acc02);
      acc12 = WMMA_BF16(a1, b2, acc12);
      acc03 = WMMA_BF16(a0, b3, acc03);
      acc13 = WMMA_BF16(a1, b3, acc13);
    }
    __syncthreads();
  }

  store_tile<OUT_MODE>(acc00, m0,      bn0 + n0w +  0, r, half, bias, add, out, N);
  store_tile<OUT_MODE>(acc01, m0,      bn0 + n0w + 16, r, half, bias, add, out, N);
  store_tile<OUT_MODE>(acc02, m0,      bn0 + n0w + 32, r, half, bias, add, out, N);
  store_tile<OUT_MODE>(acc03, m0,      bn0 + n0w + 48, r, half, bias, add, out, N);
  store_tile<OUT_MODE>(acc10, m0 + 16, bn0 + n0w +  0, r, half, bias, add, out, N);
  store_tile<OUT_MODE>(acc11, m0 + 16, bn0 + n0w + 16, r, half, bias, add, out, N);
  store_tile<OUT_MODE>(acc12, m0 + 16, bn0 + n0w + 32, r, half, bias, add, out, N);
  store_tile<OUT_MODE>(acc13, m0 + 16, bn0 + n0w + 48, r, half, bias, add, out, N);
}

// ---------- LayerNorm -> bf16 ----------
__global__ __launch_bounds__(256) void layernorm_bf16(
    const float* __restrict__ x, const float* __restrict__ g,
    const float* __restrict__ b, u16* __restrict__ h) {
  __shared__ float red[256];
  const int row = blockIdx.x;
  const int tid = threadIdx.x;
  const float* xr = x + (size_t)row * DIMC;
  float s = 0.f, ss = 0.f;
#pragma unroll
  for (int c = tid; c < DIMC; c += 256) { float v = xr[c]; s += v; ss += v * v; }
  red[tid] = s; __syncthreads();
  for (int st = 128; st > 0; st >>= 1) { if (tid < st) red[tid] += red[tid + st]; __syncthreads(); }
  float mean = red[0] / DIMC;
  __syncthreads();
  red[tid] = ss; __syncthreads();
  for (int st = 128; st > 0; st >>= 1) { if (tid < st) red[tid] += red[tid + st]; __syncthreads(); }
  float var  = red[0] / DIMC - mean * mean;
  float rstd = rsqrtf(var + EPS_LN);
  u16* hr = h + (size_t)row * DIMC;
#pragma unroll
  for (int c = tid; c < DIMC; c += 256)
    hr[c] = f2bf((xr[c] - mean) * rstd * g[c] + b[c]);
}

// ---------- L2-normalize rows of [T,256] in place; optionally A = (qn . wg)*SCALE ----------
__global__ __launch_bounds__(256) void l2norm_rows(
    float* __restrict__ q, const float* __restrict__ wg, float* __restrict__ Aout) {
  __shared__ float red[256];
  const int row = blockIdx.x;
  const int c   = threadIdx.x;
  float* qr = q + (size_t)row * ADIMC;
  float v = qr[c];
  red[c] = v * v; __syncthreads();
  for (int st = 128; st > 0; st >>= 1) { if (c < st) red[c] += red[c + st]; __syncthreads(); }
  float inv = 1.0f / fmaxf(sqrtf(red[0]), EPS_NRM);
  float vn = v * inv;
  qr[c] = vn;
  if (wg) {
    __syncthreads();
    red[c] = vn * wg[c]; __syncthreads();
    for (int st = 128; st > 0; st >>= 1) { if (c < st) red[c] += red[c + st]; __syncthreads(); }
    if (c == 0) Aout[row] = red[0] * SCALE_A;
  }
}

// ---------- normalize A over token dim per batch ----------
__global__ __launch_bounds__(256) void norm_A(float* __restrict__ A) {
  __shared__ float red[256];
  const int b = blockIdx.x, tid = threadIdx.x;
  float* Ab = A + (size_t)b * NTOK;
  float s = 0.f;
  for (int t = tid; t < NTOK; t += 256) { float a = Ab[t]; s += a * a; }
  red[tid] = s; __syncthreads();
  for (int st = 128; st > 0; st >>= 1) { if (tid < st) red[tid] += red[tid + st]; __syncthreads(); }
  float inv = 1.0f / fmaxf(sqrtf(red[0]), EPS_NRM);
  for (int t = tid; t < NTOK; t += 256) Ab[t] *= inv;
}

// ---------- G[b,c] = sum_t A[b,t] * q[b,t,c] ----------
__global__ __launch_bounds__(256) void compute_G(
    const float* __restrict__ A, const float* __restrict__ q, float* __restrict__ G) {
  const int b = blockIdx.x, c = threadIdx.x;
  const float* qb = q + (size_t)b * NTOK * ADIMC;
  const float* Ab = A + (size_t)b * NTOK;
  float acc = 0.f;
  for (int t = 0; t < NTOK; ++t) acc += Ab[t] * qb[(size_t)t * ADIMC + c];
  G[b * ADIMC + c] = acc;
}

// ---------- gk[t,c] = bf16( G[b(t),c] * k[t,c] ) ----------
__global__ __launch_bounds__(256) void make_gk(
    const float* __restrict__ G, const float* __restrict__ k, u16* __restrict__ gk) {
  size_t i = (size_t)blockIdx.x * 256 + threadIdx.x;
  if (i >= (size_t)T_TOKENS * ADIMC) return;
  int c = (int)(i & (ADIMC - 1));
  int t = (int)(i >> 8);
  int b = t >> 12;
  gk[i] = f2bf(G[b * ADIMC + c] * k[i]);
}

// ---------- W[K,N] fp32 -> Wt[N,K] bf16 ----------
__global__ __launch_bounds__(256) void transpose_bf16(
    const float* __restrict__ W, u16* __restrict__ Wt, int K, int N) {
  size_t i = (size_t)blockIdx.x * 256 + threadIdx.x;
  size_t total = (size_t)K * N;
  if (i >= total) return;
  int n = (int)(i % N);
  int k = (int)(i / N);
  Wt[(size_t)n * K + k] = f2bf(W[i]);
}

// =====================================================================
extern "C" void kernel_launch(void* const* d_in, const int* in_sizes, int n_in,
                              void* d_out, int out_size, void* d_ws, size_t ws_size,
                              hipStream_t stream) {
  (void)in_sizes; (void)n_in; (void)out_size; (void)ws_size;
  const float* x_in  = (const float*)d_in[0];
  const float* ln1_g = (const float*)d_in[1];
  const float* ln1_b = (const float*)d_in[2];
  const float* Wq    = (const float*)d_in[3];
  const float* bq    = (const float*)d_in[4];
  const float* Wk    = (const float*)d_in[5];
  const float* bk    = (const float*)d_in[6];
  const float* wg    = (const float*)d_in[7];
  const float* Wp    = (const float*)d_in[8];
  const float* bp    = (const float*)d_in[9];
  const float* Wf    = (const float*)d_in[10];
  const float* bf_   = (const float*)d_in[11];
  const float* ln2_g = (const float*)d_in[12];
  const float* ln2_b = (const float*)d_in[13];
  const float* W1    = (const float*)d_in[14];
  const float* b1    = (const float*)d_in[15];
  const float* W2    = (const float*)d_in[16];
  const float* b2    = (const float*)d_in[17];

  float* x = (float*)d_out;   // residual stream lives in d_out
  hipMemcpyAsync(x, x_in, (size_t)T_TOKENS * DIMC * sizeof(float),
                 hipMemcpyDeviceToDevice, stream);

  char* ws = (char*)d_ws;
  auto carve = [&](size_t bytes) {
    char* p = ws; ws += (bytes + 255) & ~(size_t)255; return p;
  };
  u16*   h    = (u16*)  carve((size_t)T_TOKENS * DIMC * 2);
  float* q    = (float*)carve((size_t)T_TOKENS * ADIMC * 4);
  float* k    = (float*)carve((size_t)T_TOKENS * ADIMC * 4);
  float* Abuf = (float*)carve((size_t)T_TOKENS * 4);
  float* G    = (float*)carve((size_t)NBATCH * ADIMC * 4);
  u16*   gk   = (u16*)  carve((size_t)T_TOKENS * ADIMC * 2);
  u16*   o256 = (u16*)  carve((size_t)T_TOKENS * ADIMC * 2);
  u16*   y1   = (u16*)  carve((size_t)4096 * HIDC * 2);       // MLP chunk buffer
  u16*   WqT  = (u16*)  carve((size_t)ADIMC * DIMC * 2);
  u16*   WkT  = (u16*)  carve((size_t)ADIMC * DIMC * 2);
  u16*   WpT  = (u16*)  carve((size_t)ADIMC * ADIMC * 2);
  u16*   WfT  = (u16*)  carve((size_t)DIMC * ADIMC * 2);
  u16*   W1T  = (u16*)  carve((size_t)HIDC * DIMC * 2);
  u16*   W2T  = (u16*)  carve((size_t)DIMC * HIDC * 2);

  for (int L = 0; L < 4; ++L) {
    auto tconv = [&](const float* W, u16* Wt, int Kd, int Nd) {
      size_t tot = (size_t)Kd * Nd;
      transpose_bf16<<<dim3((unsigned)((tot + 255) / 256)), dim3(256), 0, stream>>>(W, Wt, Kd, Nd);
    };
    tconv(Wq + (size_t)L * DIMC * ADIMC, WqT, DIMC, ADIMC);
    tconv(Wk + (size_t)L * DIMC * ADIMC, WkT, DIMC, ADIMC);
    tconv(Wp + (size_t)L * ADIMC * ADIMC, WpT, ADIMC, ADIMC);
    tconv(Wf + (size_t)L * ADIMC * DIMC, WfT, ADIMC, DIMC);
    tconv(W1 + (size_t)L * DIMC * HIDC, W1T, DIMC, HIDC);
    tconv(W2 + (size_t)L * HIDC * DIMC, W2T, HIDC, DIMC);

    // ---- attention branch ----
    layernorm_bf16<<<dim3(T_TOKENS), dim3(256), 0, stream>>>(
        x, ln1_g + (size_t)L * DIMC, ln1_b + (size_t)L * DIMC, h);

    gemm_bf16_wmma<0><<<dim3(ADIMC / BN, T_TOKENS / BM), dim3(256), 0, stream>>>(
        h, WqT, bq + (size_t)L * ADIMC, nullptr, q, T_TOKENS, ADIMC, DIMC);
    gemm_bf16_wmma<0><<<dim3(ADIMC / BN, T_TOKENS / BM), dim3(256), 0, stream>>>(
        h, WkT, bk + (size_t)L * ADIMC, nullptr, k, T_TOKENS, ADIMC, DIMC);

    l2norm_rows<<<dim3(T_TOKENS), dim3(256), 0, stream>>>(q, wg + (size_t)L * ADIMC, Abuf);
    l2norm_rows<<<dim3(T_TOKENS), dim3(256), 0, stream>>>(k, nullptr, nullptr);
    norm_A<<<dim3(NBATCH), dim3(256), 0, stream>>>(Abuf);
    compute_G<<<dim3(NBATCH), dim3(256), 0, stream>>>(Abuf, q, G);
    make_gk<<<dim3((T_TOKENS * ADIMC + 255) / 256), dim3(256), 0, stream>>>(G, k, gk);

    // o256 = (G*k) @ Wp + bp + q   (stored bf16 for next GEMM)
    gemm_bf16_wmma<1><<<dim3(ADIMC / BN, T_TOKENS / BM), dim3(256), 0, stream>>>(
        gk, WpT, bp + (size_t)L * ADIMC, q, o256, T_TOKENS, ADIMC, ADIMC);
    // x += o256 @ Wf + bf
    gemm_bf16_wmma<0><<<dim3(DIMC / BN, T_TOKENS / BM), dim3(256), 0, stream>>>(
        o256, WfT, bf_ + (size_t)L * DIMC, x, x, T_TOKENS, DIMC, ADIMC);

    // ---- MLP branch ----
    layernorm_bf16<<<dim3(T_TOKENS), dim3(256), 0, stream>>>(
        x, ln2_g + (size_t)L * DIMC, ln2_b + (size_t)L * DIMC, h);

    for (int c = 0; c < 4; ++c) {  // 4 token chunks of 4096 to bound workspace
      const u16* hA = h + (size_t)c * 4096 * DIMC;
      float* xc = x + (size_t)c * 4096 * DIMC;
      gemm_bf16_wmma<2><<<dim3(HIDC / BN, 4096 / BM), dim3(256), 0, stream>>>(
          hA, W1T, b1 + (size_t)L * HIDC, nullptr, y1, 4096, HIDC, DIMC);
      gemm_bf16_wmma<0><<<dim3(DIMC / BN, 4096 / BM), dim3(256), 0, stream>>>(
          y1, W2T, b2 + (size_t)L * DIMC, xc, xc, 4096, DIMC, HIDC);
    }
  }
}
